// Model_48438641164877
// MI455X (gfx1250) — matrix operand order repaired and checked by reading, not compiled, not run on hardware
//
#include <hip/hip_runtime.h>
#include <hip/hip_bf16.h>

#define DEV __device__ __forceinline__

typedef __attribute__((ext_vector_type(16))) _Float16 v16h;
typedef __attribute__((ext_vector_type(8)))  float    v8f;

static constexpr int B_ = 16, L_ = 2048, M_ = 64, H_ = 16;
static constexpr int T_ = 256, PROJ_ = 128, PRED_ = 336, HID_ = 64;

union U16h { uint4 q[2]; v16h v; };

// A-fragment (16x32 f16, MxK): lane&15 = M row; lanes<16 hold K {0..7,16..23},
// lanes>=16 hold K {8..15,24..31}  (ISA 7.12.2, 16-bit A-matrix table)
DEV v16h load_a_frag(const _Float16* base, int row_stride, int m_off, int kc, int lane) {
  int m  = m_off + (lane & 15);
  int kb = (lane >> 4) * 8;
  const _Float16* p = base + (size_t)m * row_stride + kc + kb;
  U16h u;
  u.q[0] = *(const uint4*)(p);        // K = kc+kb .. +7
  u.q[1] = *(const uint4*)(p + 16);   // K = kc+16+kb .. +7
  return u.v;
}

// B-fragment (32x16 f16, KxN): lane&15 = N col; lanes<16 hold K {0..7,16..23},
// lanes>=16 hold K {8..15,24..31} (same K order as A)
DEV v16h load_b_frag(const _Float16* base, int row_stride, int kc, int n_off, int lane) {
  int n  = n_off + (lane & 15);
  int kb = (lane >> 4) * 8;
  v16h b;
  for (int s = 0; s < 8; ++s) {
    b[s]     = base[(size_t)(kc + kb + s) * row_stride + n];
    b[8 + s] = base[(size_t)(kc + 16 + kb + s) * row_stride + n];
  }
  return b;
}

DEV v8f wmma16(v16h a, v16h b, v8f c) {
  return __builtin_amdgcn_wmma_f32_16x16x32_f16(false, a, false, b, (short)0, c, false, false);
}

DEV float gelu_tanh(float x) {
  float x3 = x * x * x;
  return 0.5f * x * (1.f + tanhf(0.7978845608028654f * (x + 0.044715f * x3)));
}
DEV float sigmoidf(float x) { return 1.f / (1.f + __expf(-x)); }

// async copy of 43008 bytes (64 K-rows x 336 cols f16) global -> LDS.
// Each of the 256 threads issues 21 x b64 copies => 21 async ops per wave.
DEV void issue_async_half(_Float16* dstLds, const _Float16* srcGlobal, int tid) {
  unsigned lbase = (unsigned)(size_t)dstLds;            // low 32 bits = LDS offset
  const char* src = (const char*)srcGlobal;
  for (int r = 0; r < 21; ++r) {
    unsigned off = (unsigned)(r * 256 + tid) * 8u;
    unsigned laddr = lbase + off;
    const char* g = src + off;
    asm volatile("global_load_async_to_lds_b64 %0, %1, off"
                 :: "v"(laddr), "v"(g) : "memory");
  }
}

// ---------------- f32 -> f16 convert ----------------
__global__ void cvt_f16_kernel(const float* __restrict__ src, _Float16* __restrict__ dst, int n) {
  int i = blockIdx.x * blockDim.x + threadIdx.x;
  if (i < n) dst[i] = (_Float16)src[i];
}

// ---------------- stats = mean over L (2-stage) ----------------
__global__ void stats1_kernel(const float* __restrict__ x, float* __restrict__ part) {
  __shared__ float red[4][64];
  int blk = blockIdx.x, b = blk >> 4, lc = blk & 15;     // 16 l-chunks of 128
  int tid = threadIdx.x, m = tid & 63, g = tid >> 6;
  const float* p = x + (size_t)b * L_ * M_ + (size_t)(lc * 128 + g * 32) * M_ + m;
  float s = 0.f;
  for (int l = 0; l < 32; ++l) s += p[(size_t)l * M_];
  red[g][m] = s;
  __syncthreads();
  if (g == 0) part[blk * 64 + m] = red[0][m] + red[1][m] + red[2][m] + red[3][m];
}
__global__ void stats2_kernel(const float* __restrict__ part, float* __restrict__ stats) {
  int gid = blockIdx.x * blockDim.x + threadIdx.x;       // 0..1023
  int b = gid >> 6, m = gid & 63;
  float s = 0.f;
  for (int lc = 0; lc < 16; ++lc) s += part[(b * 16 + lc) * 64 + m];
  stats[gid] = s * (1.0f / L_);
}

// ---------------- SACA branch: [64]->gelu[64]->[1024] ----------------
__global__ void saca_kernel(const float* __restrict__ stats, const float* __restrict__ W1,
                            const float* __restrict__ b1, const float* __restrict__ W2,
                            const float* __restrict__ b2, float* __restrict__ out, int do_sig) {
  __shared__ float s[64], h[64];
  int b = blockIdx.x, tid = threadIdx.x;
  if (tid < 64) s[tid] = stats[b * 64 + tid];
  __syncthreads();
  if (tid < 64) {
    float acc = b1[tid];
    for (int i = 0; i < 64; ++i) acc += s[i] * W1[i * 64 + tid];
    h[tid] = gelu_tanh(acc);
  }
  __syncthreads();
  for (int n = tid; n < 1024; n += 256) {
    float acc = b2 ? b2[n] : 0.f;
    for (int i = 0; i < 64; ++i) acc += h[i] * W2[i * 1024 + n];
    out[b * 1024 + n] = do_sig ? sigmoidf(acc) : acc;
  }
}

// ---------------- patch embed -> z[B,H,M,T] f32 ----------------
__global__ void embed_kernel(const float* __restrict__ x, const float* __restrict__ We,
                             const float* __restrict__ be, float* __restrict__ z) {
  __shared__ float w[16][16];
  __shared__ float bias[16];
  int tid = threadIdx.x;
  w[tid >> 4][tid & 15] = We[tid];
  if (tid < 16) bias[tid] = be[tid];
  __syncthreads();
  int bm = blockIdx.x, b = bm >> 6, m = bm & 63;
  int t = tid;
  float xs[16];
  for (int p = 0; p < 16; ++p) {
    int pos = t * 8 + p - 8;               // front pad = 8, repeat last value
    int l = (pos < 0) ? (L_ - 1) : pos;
    xs[p] = x[(size_t)b * L_ * M_ + (size_t)l * M_ + m];
  }
  for (int hh = 0; hh < 16; ++hh) {
    float acc = bias[hh];
    for (int p = 0; p < 16; ++p) acc += xs[p] * w[p][hh];
    z[(((size_t)b * 16 + hh) * 64 + m) * 256 + t] = acc;
  }
}

// ---------------- channel attention per (b,h), WMMA ----------------
__global__ void attn_kernel(float* __restrict__ z,
                            const float* __restrict__ sigq, const float* __restrict__ sigk,
                            const float* __restrict__ sigv, const float* __restrict__ muq,
                            const float* __restrict__ muk) {
  __shared__ _Float16 qf[64][256];   // q row-major (A for scores)
  __shared__ _Float16 kT[256][64];   // k transposed (B for scores: B[t][n]=k[n][t])
  __shared__ _Float16 vv[64][256];   // v row-major (B for out: B[n][t])
  __shared__ float    sc[64][72];    // scores f32 (padded)
  __shared__ _Float16 at[64][64];    // softmaxed attn f16
  __shared__ float sq[64], sk[64], sv[64], mq[64], mk[64];

  int bh = blockIdx.x, b = bh >> 4, hh = bh & 15;
  int tid = threadIdx.x;
  float* zb = z + (size_t)bh * 64 * 256;
  if (tid < 64) {
    int m = tid;
    sq[m] = sigq[b * 1024 + m * 16 + hh];
    sk[m] = sigk[b * 1024 + m * 16 + hh];
    sv[m] = sigv[b * 1024 + m * 16 + hh];
    mq[m] = muq[b * 1024 + m * 16 + hh];
    mk[m] = muk[b * 1024 + m * 16 + hh];
  }
  __syncthreads();
  for (int idx = tid; idx < 64 * 256; idx += 256) {
    int m = idx >> 8, t = idx & 255;
    float zv = zb[m * 256 + t];
    qf[m][t] = (_Float16)(zv * sq[m] + mq[m]);
    kT[t][m] = (_Float16)(zv * sk[m] + mk[m]);
    vv[m][t] = (_Float16)(zv * sv[m]);
  }
  __syncthreads();

  int wave = tid >> 5, lane = tid & 31;
  // scores: 4x4 tiles of 16x16, K=256; wave handles 2 tiles
  for (int ti = 0; ti < 2; ++ti) {
    int tile = wave * 2 + ti, mt = tile >> 2, nt = tile & 3;
    v8f acc = {};
    for (int kc = 0; kc < 256; kc += 32) {
      v16h a = load_a_frag(&qf[0][0], 256, mt * 16, kc, lane);
      v16h bb = load_b_frag(&kT[0][0], 64, kc, nt * 16, lane);
      acc = wmma16(a, bb, acc);
    }
    int n = lane & 15, mb = (lane >> 4) * 8;
    for (int r = 0; r < 8; ++r)
      sc[mt * 16 + mb + r][nt * 16 + n] = acc[r] * 0.0625f;  // 1/sqrt(T)
  }
  __syncthreads();
  if (tid < 64) {
    int m = tid;
    float mx = -1e30f;
    for (int n = 0; n < 64; ++n) mx = fmaxf(mx, sc[m][n]);
    float s = 0.f;
    for (int n = 0; n < 64; ++n) s += __expf(sc[m][n] - mx);
    float inv = 1.f / s;
    for (int n = 0; n < 64; ++n) at[m][n] = (_Float16)(__expf(sc[m][n] - mx) * inv);
  }
  __syncthreads();
  // out = attn(64x64) @ v(64x256), residual preloaded in C; 64 tiles / 8 waves
  for (int i = 0; i < 8; ++i) {
    int id = wave * 8 + i, mt = id >> 4, tt = id & 15;
    int n = lane & 15, mb = (lane >> 4) * 8;
    v8f acc;
    for (int r = 0; r < 8; ++r)
      acc[r] = zb[(mt * 16 + mb + r) * 256 + tt * 16 + n];   // residual
    for (int kc = 0; kc < 64; kc += 32) {
      v16h a = load_a_frag(&at[0][0], 64, mt * 16, kc, lane);
      v16h bb = load_b_frag(&vv[0][0], 256, kc, tt * 16, lane);
      acc = wmma16(a, bb, acc);
    }
    for (int r = 0; r < 8; ++r)
      zb[(mt * 16 + mb + r) * 256 + tt * 16 + n] = acc[r];
  }
}

// ---------------- FFN over token dim, 16 rows per block, WMMA ----------------
__global__ void ffn_kernel(float* __restrict__ z, const _Float16* __restrict__ Wf1h,
                           const _Float16* __restrict__ Wf2h, const float* __restrict__ bf1,
                           const float* __restrict__ bf2) {
  __shared__ _Float16 zt[16][256];
  __shared__ _Float16 Yt[16][1024];
  __shared__ float b1s[1024];
  int tid = threadIdx.x;
  float* zrow = z + (size_t)blockIdx.x * 16 * 256;
  for (int idx = tid; idx < 4096; idx += 256) {
    int i = idx >> 8, t = idx & 255;
    zt[i][t] = (_Float16)zrow[i * 256 + t];
  }
  for (int n = tid; n < 1024; n += 256) b1s[n] = bf1[n];
  __syncthreads();

  int wave = tid >> 5, lane = tid & 31;
  int n0 = lane & 15, mb = (lane >> 4) * 8;
  // GEMM1: [16,256] @ [256,1024], 64 N-tiles / 8 waves
  {
    v8f acc[8] = {};
    for (int kc = 0; kc < 256; kc += 32) {
      v16h a = load_a_frag(&zt[0][0], 256, 0, kc, lane);
      for (int i = 0; i < 8; ++i) {
        v16h bb = load_b_frag(Wf1h, 1024, kc, (wave * 8 + i) * 16, lane);
        acc[i] = wmma16(a, bb, acc[i]);
      }
    }
    for (int i = 0; i < 8; ++i) {
      int nt = wave * 8 + i;
      for (int r = 0; r < 8; ++r) {
        float v = acc[i][r] + b1s[nt * 16 + n0];
        Yt[mb + r][nt * 16 + n0] = (_Float16)gelu_tanh(v);
      }
    }
  }
  __syncthreads();
  // GEMM2: [16,1024] @ [1024,256], 16 N-tiles / 8 waves (2 each), + bias + residual
  {
    v8f acc[2] = {};
    for (int kc = 0; kc < 1024; kc += 32) {
      v16h a = load_a_frag(&Yt[0][0], 1024, 0, kc, lane);
      for (int i = 0; i < 2; ++i) {
        v16h bb = load_b_frag(Wf2h, 256, kc, (wave * 2 + i) * 16, lane);
        acc[i] = wmma16(a, bb, acc[i]);
      }
    }
    for (int i = 0; i < 2; ++i) {
      int nt = wave * 2 + i;
      for (int r = 0; r < 8; ++r) {
        int m = mb + r, n = nt * 16 + n0;
        zrow[m * 256 + n] = zrow[m * 256 + n] + acc[i][r] + bf2[n];
      }
    }
  }
}

// ------- fused head: proj(H->128)+Swish on the fly, FC K=32768 via WMMA.
// K split 4-way over blocks; Wfc staged to LDS with double-buffered async copies.
__global__ void head_kernel(const float* __restrict__ z, const float* __restrict__ Wp,
                            const float* __restrict__ bp, const _Float16* __restrict__ Wfch,
                            float* __restrict__ partial) {
  __shared__ float Wps[16][128];
  __shared__ float bps[128];
  __shared__ float zsl[16][16];
  __shared__ _Float16 aCh[16][128];
  __shared__ __align__(32) _Float16 wbuf[2][64 * 336];   // 2 x 42KB half-chunks

  int tid = threadIdx.x;
  int rowblk = blockIdx.x & 63;        // 16-row tile of the 1024 (b,m) rows
  int kc4 = blockIdx.x >> 6;           // 0..3 : K-split over token range
  int t0 = kc4 * 64;
  for (int i = tid; i < 2048; i += 256) Wps[i >> 7][i & 127] = Wp[i];
  if (tid < 128) bps[tid] = bp[tid];
  int row0 = rowblk * 16;
  int wave = tid >> 5, lane = tid & 31;
  int i16 = tid >> 4, j16 = tid & 15;
  v8f acc[3] = {};
  __syncthreads();

  // slot s (0..127): t = t0 + s/2, K-rows (s&1)*64 .. +63 of that token's chunk
  issue_async_half(&wbuf[0][0], Wfch + (size_t)t0 * 128 * 336, tid);

  for (int ti = 0; ti < 64; ++ti) {
    int t = t0 + ti;
    {  // gather z slice [16 rows][16 heads] at token t
      int r = row0 + i16, b = r >> 6, m = r & 63;
      zsl[i16][j16] = z[(((size_t)b * 16 + j16) * 64 + m) * 256 + t];
    }
    __syncthreads();
    {  // proj + Swish -> f16 A-chunk [16][128]
      int pb = j16 * 8;
      for (int pp = 0; pp < 8; ++pp) {
        int p = pb + pp;
        float s = bps[p];
        for (int hh = 0; hh < 16; ++hh) s += zsl[i16][hh] * Wps[hh][p];
        aCh[i16][p] = (_Float16)(s * sigmoidf(s));  // Swish
      }
    }
    __syncthreads();
    for (int h = 0; h < 2; ++h) {
      int slot = ti * 2 + h;
      if (slot + 1 < 128) {
        int ns = slot + 1;
        issue_async_half(&wbuf[ns & 1][0],
                         Wfch + ((size_t)(t0 + (ns >> 1)) * 128 + (ns & 1) * 64) * 336, tid);
        // async-LDS loads complete in order: <=21 outstanding => current half resident
        asm volatile("s_wait_asynccnt 0x15" ::: "memory");
      } else {
        asm volatile("s_wait_asynccnt 0x0" ::: "memory");
      }
      __syncthreads();   // all waves' copies visible before consuming
      const _Float16* wb = &wbuf[slot & 1][0];
      for (int kc2 = 0; kc2 < 64; kc2 += 32) {
        v16h a = load_a_frag(&aCh[0][0], 128, 0, h * 64 + kc2, lane);
        for (int i = 0; i < 3; ++i) {
          int nt = wave * 3 + i;          // wave-uniform -> EXEC stays all-ones
          if (nt < 21) {
            v16h bb = load_b_frag(wb, 336, kc2, nt * 16, lane);
            acc[i] = wmma16(a, bb, acc[i]);
          }
        }
      }
      __syncthreads();   // everyone done reading before next prefetch overwrites
    }
  }
  int n0 = lane & 15, mb = (lane >> 4) * 8;
  for (int i = 0; i < 3; ++i) {
    int nt = wave * 3 + i;
    if (nt < 21) {
      for (int r = 0; r < 8; ++r) {
        int n = nt * 16 + n0;
        partial[(((size_t)kc4 * 1024) + row0 + mb + r) * 336 + n] = acc[i][r];
      }
    }
  }
}

// deterministic K-split reduction + bias
__global__ void head_reduce_kernel(const float* __restrict__ partial,
                                   const float* __restrict__ bfc, float* __restrict__ out) {
  int i = blockIdx.x * blockDim.x + threadIdx.x;
  if (i < 1024 * 336) {
    int n = i % 336;
    float s = partial[i] + partial[344064 + i] + partial[2 * 344064 + i] + partial[3 * 344064 + i];
    out[i] = s + bfc[n];
  }
}

// ---------------- host launcher ----------------
extern "C" void kernel_launch(void* const* d_in, const int* in_sizes, int n_in,
                              void* d_out, int out_size, void* d_ws, size_t ws_size,
                              hipStream_t stream) {
  const float* x      = (const float*)d_in[0];
  const float* W_emb  = (const float*)d_in[1];
  const float* b_emb  = (const float*)d_in[2];
  const float* Wsq1 = (const float*)d_in[3],  *bsq1 = (const float*)d_in[4];
  const float* Wsq2 = (const float*)d_in[5],  *bsq2 = (const float*)d_in[6];
  const float* Wsk1 = (const float*)d_in[7],  *bsk1 = (const float*)d_in[8];
  const float* Wsk2 = (const float*)d_in[9],  *bsk2 = (const float*)d_in[10];
  const float* Wsv1 = (const float*)d_in[11], *bsv1 = (const float*)d_in[12];
  const float* Wsv2 = (const float*)d_in[13], *bsv2 = (const float*)d_in[14];
  const float* Wmq1 = (const float*)d_in[15], *bmq1 = (const float*)d_in[16];
  const float* Wmq2 = (const float*)d_in[17];
  const float* Wmk1 = (const float*)d_in[18], *bmk1 = (const float*)d_in[19];
  const float* Wmk2 = (const float*)d_in[20];
  const float* Wf1  = (const float*)d_in[21], *bf1 = (const float*)d_in[22];
  const float* Wf2  = (const float*)d_in[23], *bf2 = (const float*)d_in[24];
  const float* Wp   = (const float*)d_in[25], *bp  = (const float*)d_in[26];
  const float* Wfc  = (const float*)d_in[27], *bfc = (const float*)d_in[28];
  float* out = (float*)d_out;

  char* ws = (char*)d_ws;
  // workspace layout (all 256B aligned)
  float*    stats = (float*)(ws + 0);                  //   4 KB
  float*    saca0 = (float*)(ws + 4096);               // 5 * 64 KB
  float*    sigq  = saca0 + 0 * 16384;
  float*    sigk  = saca0 + 1 * 16384;
  float*    sigv  = saca0 + 2 * 16384;
  float*    muq   = saca0 + 3 * 16384;
  float*    muk   = saca0 + 4 * 16384;
  float*    z     = (float*)(ws + 331776);             // 16 MB  [B,H,M,T]
  _Float16* Wf1h  = (_Float16*)(ws + 17108992);        // 512 KB [256,1024]
  _Float16* Wf2h  = (_Float16*)(ws + 17633280);        // 512 KB [1024,256]
  _Float16* Wfch  = (_Float16*)(ws + 18157568);        //  21 MB [32768,336]
  float*    hpart = (float*)(ws + 40177664);           // 5.25 MB [4][1024][336]
  float*    spart = (float*)(ws + 45682688);           //  64 KB [256][64]
  // total ~43.6 MB of workspace

  // weight conversions to f16
  cvt_f16_kernel<<<(262144 + 255) / 256, 256, 0, stream>>>(Wf1, Wf1h, 262144);
  cvt_f16_kernel<<<(262144 + 255) / 256, 256, 0, stream>>>(Wf2, Wf2h, 262144);
  cvt_f16_kernel<<<(11010048 + 255) / 256, 256, 0, stream>>>(Wfc, Wfch, 11010048);

  stats1_kernel<<<256, 256, 0, stream>>>(x, spart);
  stats2_kernel<<<4, 256, 0, stream>>>(spart, stats);

  saca_kernel<<<16, 256, 0, stream>>>(stats, Wsq1, bsq1, Wsq2, bsq2, sigq, 1);
  saca_kernel<<<16, 256, 0, stream>>>(stats, Wsk1, bsk1, Wsk2, bsk2, sigk, 1);
  saca_kernel<<<16, 256, 0, stream>>>(stats, Wsv1, bsv1, Wsv2, bsv2, sigv, 1);
  saca_kernel<<<16, 256, 0, stream>>>(stats, Wmq1, bmq1, Wmq2, nullptr, muq, 0);
  saca_kernel<<<16, 256, 0, stream>>>(stats, Wmk1, bmk1, Wmk2, nullptr, muk, 0);

  embed_kernel<<<B_ * M_, 256, 0, stream>>>(x, W_emb, b_emb, z);
  attn_kernel<<<B_ * H_, 256, 0, stream>>>(z, sigq, sigk, sigv, muq, muk);
  ffn_kernel<<<(B_ * H_ * M_) / 16, 256, 0, stream>>>(z, Wf1h, Wf2h, bf1, bf2);
  head_kernel<<<256, 256, 0, stream>>>(z, Wp, bp, Wfch, hpart);
  head_reduce_kernel<<<(1024 * 336 + 255) / 256, 256, 0, stream>>>(hpart, bfc, out);

  (void)in_sizes; (void)n_in; (void)out_size; (void)ws_size;
}
